// HypGRU_4346506903716
// MI455X (gfx1250) — compile-verified, matrix operand order checked by reading, not run on hardware
//
#include <hip/hip_runtime.h>
#include <hip/hip_bf16.h>

// ---------------------------------------------------------------------------
// HypGRU (hyperbolic GRU cell) for MI455X / gfx1250.
// N=16384, H=1024.  6 GEMMs (~206 GFLOP) done in bf16 WMMA w/ f32 accum;
// global->LDS staging via gfx1250 async-to-LDS DMA (double-buffered) when the
// toolchain exposes the builtins; Mobius / logmap0 / sigmoid row-ops done as
// per-row block-reduction kernels.
// ---------------------------------------------------------------------------

typedef __bf16 bf16_t;
typedef bf16_t v16bf __attribute__((ext_vector_type(16)));
typedef float  v8f   __attribute__((ext_vector_type(8)));
typedef unsigned int v4u __attribute__((ext_vector_type(4)));
typedef unsigned int v8u __attribute__((ext_vector_type(8)));

#define HDIM 1024
#define NROWS 16384

#if __has_builtin(__builtin_amdgcn_global_load_async_to_lds_b128)
#define USE_ASYNC_LDS 1
#else
#define USE_ASYNC_LDS 0
#endif

// ---------------- fp32 <-> bf16 helpers (round-to-nearest-even) ------------
__device__ __forceinline__ unsigned short f32_to_bf16(float f) {
    unsigned int u = __float_as_uint(f);
    unsigned int rnd = 0x7FFFu + ((u >> 16) & 1u);
    u += rnd;
    return (unsigned short)(u >> 16);
}
__device__ __forceinline__ float bf16_to_f32(unsigned short h) {
    return __uint_as_float(((unsigned int)h) << 16);
}

__global__ void cvt_bf16_kernel(const float* __restrict__ in,
                                unsigned short* __restrict__ out, int n) {
    int i = blockIdx.x * blockDim.x + threadIdx.x;
    int stride = gridDim.x * blockDim.x;
    for (; i < n; i += stride) out[i] = f32_to_bf16(in[i]);
}

#if USE_ASYNC_LDS
// builtin expects int4 (vector_size 16) pointers: arg0 = global (as1),
// arg1 = LDS (as3), then imm offset + imm cpol.
typedef int v4i_vs __attribute__((vector_size(16)));
typedef v4i_vs __attribute__((address_space(1))) as1_v4i;
typedef v4i_vs __attribute__((address_space(3))) as3_v4i;

__device__ __forceinline__ void async_b128(const unsigned short* g,
                                           unsigned short* l) {
    // global (as1) -> LDS (as3) async DMA, tracked by ASYNCcnt
    __builtin_amdgcn_global_load_async_to_lds_b128(
        (as1_v4i*)(unsigned long long)g,
        (as3_v4i*)(unsigned int)(unsigned long long)l,
        0, 0);
}
template <int N>
__device__ __forceinline__ void wait_async() {
#if __has_builtin(__builtin_amdgcn_s_wait_asynccnt)
    __builtin_amdgcn_s_wait_asynccnt(N);
#else
    asm volatile("s_wait_asynccnt %0" ::"i"(N) : "memory");
#endif
}
#endif

// ---------------- WMMA GEMM:  C[n,j] = sum_k A[n,k] * W[j,k] ----------------
// A: (NROWS x K) bf16 row-major, W: (J x K) bf16 row-major (weight rows),
// C: (NROWS x ldc) fp32.  Block = 128x128 output, 8 waves, wave = 32x64.
#define TILE_K 32
#define LDSS 40  // padded row stride in halves: 80B, 16B aligned, bank-clean

__global__ __launch_bounds__(256)
void gemm_bf16_wmma(const unsigned short* __restrict__ A,
                    const unsigned short* __restrict__ W,
                    float* __restrict__ C, int K, int ldc) {
#if USE_ASYNC_LDS
    __shared__ unsigned short As[2][128 * LDSS];
    __shared__ unsigned short Ws[2][128 * LDSS];
#else
    __shared__ unsigned short As[1][128 * LDSS];
    __shared__ unsigned short Ws[1][128 * LDSS];
#endif

    const int tid = threadIdx.x;
    const int rowBase = blockIdx.y * 128;
    const int colBase = blockIdx.x * 128;
    const int lane = tid & 31, wid = tid >> 5;
    const int wm = wid & 3;       // 4 waves along M (4 * 32 = 128 rows)
    const int wn = wid >> 2;      // 2 waves along N (2 * 64 = 128 cols)
    const int half = lane >> 4, lr = lane & 15;
    const int k0 = half * 8;      // 16-bit A/B fragment layout: half selects K octet

    v8f acc[2][4];
#pragma unroll
    for (int mt = 0; mt < 2; ++mt)
#pragma unroll
        for (int nt = 0; nt < 4; ++nt)
#pragma unroll
            for (int e = 0; e < 8; ++e) acc[mt][nt][e] = 0.0f;

    // staging layout: 512 16B-chunks per tile, 2 per thread per tile
    const int r0 = tid >> 2, q = tid & 3;
    const int r1 = r0 + 64;
    const int l0 = r0 * LDSS + q * 8, l1 = r1 * LDSS + q * 8;
    const int nTiles = K / TILE_K;

#if USE_ASYNC_LDS
    const unsigned short* gA0 = A + (size_t)(rowBase + r0) * K + q * 8;
    const unsigned short* gA1 = A + (size_t)(rowBase + r1) * K + q * 8;
    const unsigned short* gW0 = W + (size_t)(colBase + r0) * K + q * 8;
    const unsigned short* gW1 = W + (size_t)(colBase + r1) * K + q * 8;
    // prologue: DMA tile 0 into buffer 0
    async_b128(gA0, &As[0][l0]);
    async_b128(gA1, &As[0][l1]);
    async_b128(gW0, &Ws[0][l0]);
    async_b128(gW1, &Ws[0][l1]);
#endif

    for (int t = 0; t < nTiles; ++t) {
        const int kb = t * TILE_K;
#if USE_ASYNC_LDS
        const int cur = t & 1, nxt = cur ^ 1;
        __syncthreads();  // all waves finished reading buf[nxt] (tile t-1)
        if (t + 1 < nTiles) {
            const int ko = kb + TILE_K;
            async_b128(gA0 + ko, &As[nxt][l0]);
            async_b128(gA1 + ko, &As[nxt][l1]);
            async_b128(gW0 + ko, &Ws[nxt][l0]);
            async_b128(gW1 + ko, &Ws[nxt][l1]);
            wait_async<4>();  // tile t's 4 DMAs done (in-order completion)
        } else {
            wait_async<0>();
        }
        __syncthreads();  // tile t visible to all waves
#else
        const int cur = 0;
        const unsigned short* ap0 = A + (size_t)(rowBase + r0) * K + kb + q * 8;
        const unsigned short* ap1 = A + (size_t)(rowBase + r1) * K + kb + q * 8;
        const unsigned short* wp0 = W + (size_t)(colBase + r0) * K + kb + q * 8;
        const unsigned short* wp1 = W + (size_t)(colBase + r1) * K + kb + q * 8;
        v4u a0 = *(const v4u*)ap0;
        v4u a1 = *(const v4u*)ap1;
        v4u w0 = *(const v4u*)wp0;
        v4u w1 = *(const v4u*)wp1;
        if (kb + TILE_K < K) {  // prefetch next K-tile into caches
            __builtin_prefetch(ap0 + TILE_K, 0, 1);
            __builtin_prefetch(wp0 + TILE_K, 0, 1);
        }
        __syncthreads();  // previous compute done
        *(v4u*)&As[0][l0] = a0;
        *(v4u*)&As[0][l1] = a1;
        *(v4u*)&Ws[0][l0] = w0;
        *(v4u*)&Ws[0][l1] = w1;
        __syncthreads();  // tile visible
#endif

        v16bf af[2], bfm[4];
#pragma unroll
        for (int mt = 0; mt < 2; ++mt) {
            int ar = wm * 32 + mt * 16 + lr;
            v4u lo = *(const v4u*)&As[cur][ar * LDSS + k0];       // K = k0..k0+7
            v4u hi = *(const v4u*)&As[cur][ar * LDSS + 16 + k0];  // K = 16+k0..
            v8u p;
#pragma unroll
            for (int j = 0; j < 4; ++j) { p[j] = lo[j]; p[4 + j] = hi[j]; }
            af[mt] = __builtin_bit_cast(v16bf, p);
        }
#pragma unroll
        for (int nt = 0; nt < 4; ++nt) {
            int wr = wn * 64 + nt * 16 + lr;
            v4u lo = *(const v4u*)&Ws[cur][wr * LDSS + k0];
            v4u hi = *(const v4u*)&Ws[cur][wr * LDSS + 16 + k0];
            v8u p;
#pragma unroll
            for (int j = 0; j < 4; ++j) { p[j] = lo[j]; p[4 + j] = hi[j]; }
            bfm[nt] = __builtin_bit_cast(v16bf, p);
        }
#pragma unroll
        for (int mt = 0; mt < 2; ++mt)
#pragma unroll
            for (int nt = 0; nt < 4; ++nt)
                acc[mt][nt] = __builtin_amdgcn_wmma_f32_16x16x32_bf16(
                    false, af[mt], false, bfm[nt], (short)0, acc[mt][nt],
                    false, false);
    }

    // epilogue: 32-bit C/D layout -> lane%16 = N col, VGPR e = M row (+8 hi half)
#pragma unroll
    for (int mt = 0; mt < 2; ++mt)
#pragma unroll
        for (int nt = 0; nt < 4; ++nt) {
            int gcol = colBase + wn * 64 + nt * 16 + lr;
#pragma unroll
            for (int e = 0; e < 8; ++e) {
                int grow = rowBase + wm * 32 + mt * 16 + half * 8 + e;
                C[(size_t)grow * ldc + gcol] = acc[mt][nt][e];
            }
        }
}

// ---------------- block reductions (256 thr, wave32 shuffles + LDS) ---------
template <int NV>
__device__ __forceinline__ void breduce(float* v, float* sdata) {
    const int lane = threadIdx.x & 31;
    const int warp = threadIdx.x >> 5;
#pragma unroll
    for (int c = 0; c < NV; ++c) {
        float x = v[c];
#pragma unroll
        for (int off = 16; off > 0; off >>= 1) x += __shfl_down(x, off, 32);
        if (lane == 0) sdata[warp * NV + c] = x;
    }
    __syncthreads();
    if (threadIdx.x == 0) {
#pragma unroll
        for (int c = 0; c < NV; ++c) {
            float acc = 0.f;
            for (int w = 0; w < 8; ++w) acc += sdata[w * NV + c];
            sdata[c] = acc;
        }
    }
    __syncthreads();
#pragma unroll
    for (int c = 0; c < NV; ++c) v[c] = sdata[c];
    __syncthreads();
}

__device__ __forceinline__ float artanh_c(float x) {
    x = fminf(fmaxf(x, -0.99999f), 0.99999f);  // ATANH_CLAMP = 1 - 1e-5
    return 0.5f * (log1pf(x) - log1pf(-x));
}

// ---------------- gate kernel: sigmoid(logmap0(one_rnn_transform(...))) -----
// mode 0: write z_t (fp32).  mode 1: write rh_t = mobius_pointwise_mul(r,h) bf16.
__global__ __launch_bounds__(256)
void gate_kernel(const float* __restrict__ Gwh, const float* __restrict__ Gux,
                 const float* __restrict__ hvec, const float* __restrict__ xvec,
                 const float* __restrict__ bias, float* __restrict__ zout,
                 unsigned short* __restrict__ rhout, int mode) {
    __shared__ float sdata[64];
    const int row = blockIdx.x, t = threadIdx.x;
    const size_t base = (size_t)row * HDIM;

    float wh[4], ux[4], hv[4], xv[4], bv[4];
#pragma unroll
    for (int i = 0; i < 4; ++i) {
        int c = t + i * 256;
        wh[i] = Gwh[base + c]; ux[i] = Gux[base + c];
        hv[i] = hvec[base + c]; xv[i] = xvec[base + c];
        bv[i] = bias[c];
    }
    float s[4] = {0.f, 0.f, 0.f, 0.f};
#pragma unroll
    for (int i = 0; i < 4; ++i) {
        s[0] += wh[i] * wh[i]; s[1] += hv[i] * hv[i];
        s[2] += ux[i] * ux[i]; s[3] += xv[i] * xv[i];
    }
    breduce<4>(s, sdata);
    const float swh = s[0], sh = s[1], sux = s[2], sx = s[3];
    const float nW = fmaxf(sqrtf(swh), 1e-15f), nh = fmaxf(sqrtf(sh), 1e-15f);
    const float nU = fmaxf(sqrtf(sux), 1e-15f), nx = fmaxf(sqrtf(sx), 1e-15f);
    const float fW = (swh == 0.f) ? 0.f : tanhf(nW / nh * artanh_c(nh)) / nW;
    const float fU = (sux == 0.f) ? 0.f : tanhf(nU / nx * artanh_c(nx)) / nU;

    float Wv[4], Uv[4];
    float pxy[1] = {0.f};
#pragma unroll
    for (int i = 0; i < 4; ++i) {
        Wv[i] = fW * wh[i]; Uv[i] = fU * ux[i];
        pxy[0] += Wv[i] * Uv[i];
    }
    breduce<1>(pxy, sdata);
    const float x2 = fW * fW * swh, y2 = fU * fU * sux, xy = pxy[0];
    float den = fmaxf(1.f + 2.f * xy + x2 * y2, 1e-15f);
    float ca = (1.f + 2.f * xy + y2) / den, cb = (1.f - x2) / den;

    float t1[4], sb[3] = {0.f, 0.f, 0.f};
#pragma unroll
    for (int i = 0; i < 4; ++i) {
        t1[i] = ca * Wv[i] + cb * Uv[i];
        sb[0] += t1[i] * t1[i]; sb[1] += bv[i] * bv[i]; sb[2] += t1[i] * bv[i];
    }
    breduce<3>(sb, sdata);
    den = fmaxf(1.f + 2.f * sb[2] + sb[0] * sb[1], 1e-15f);
    ca = (1.f + 2.f * sb[2] + sb[1]) / den; cb = (1.f - sb[0]) / den;

    float t2[4], sn[1] = {0.f};
#pragma unroll
    for (int i = 0; i < 4; ++i) {
        t2[i] = ca * t1[i] + cb * bv[i];
        sn[0] += t2[i] * t2[i];
    }
    breduce<1>(sn, sdata);
    const float n2 = fmaxf(sqrtf(sn[0]), 1e-15f);
    const float fac = artanh_c(n2) / n2;

    float r[4];
#pragma unroll
    for (int i = 0; i < 4; ++i) {
        float g = fac * t2[i];
        r[i] = 1.f / (1.f + expf(-g));
    }
    if (mode == 0) {
#pragma unroll
        for (int i = 0; i < 4; ++i) zout[base + t + i * 256] = r[i];
    } else {
        float m[4], sm[1] = {0.f};
#pragma unroll
        for (int i = 0; i < 4; ++i) { m[i] = r[i] * hv[i]; sm[0] += m[i] * m[i]; }
        breduce<1>(sm, sdata);
        const float nm = fmaxf(sqrtf(sm[0]), 1e-15f);
        const float fM = (sm[0] == 0.f) ? 0.f : tanhf(nm / nh * artanh_c(nh)) / nm;
#pragma unroll
        for (int i = 0; i < 4; ++i)
            rhout[base + t + i * 256] = f32_to_bf16(fM * m[i]);
    }
}

// ---------------- final kernel: h_tilde, delta_h, h_out ---------------------
__global__ __launch_bounds__(256)
void final_kernel(const float* __restrict__ Gwh, const float* __restrict__ Gux,
                  const unsigned short* __restrict__ rh16,
                  const float* __restrict__ xvec, const float* __restrict__ hvec,
                  const float* __restrict__ zt, const float* __restrict__ bias,
                  float* __restrict__ out) {
    __shared__ float sdata[64];
    const int row = blockIdx.x, t = threadIdx.x;
    const size_t base = (size_t)row * HDIM;

    float g0[4], g1[4], rh[4], xv[4], hv[4], zv[4], bv[4];
#pragma unroll
    for (int i = 0; i < 4; ++i) {
        int c = t + i * 256;
        g0[i] = Gwh[base + c]; g1[i] = Gux[base + c];
        rh[i] = bf16_to_f32(rh16[base + c]);
        xv[i] = xvec[base + c]; hv[i] = hvec[base + c];
        zv[i] = zt[base + c]; bv[i] = bias[c];
    }
    float s[5] = {0.f, 0.f, 0.f, 0.f, 0.f};
#pragma unroll
    for (int i = 0; i < 4; ++i) {
        s[0] += g0[i] * g0[i]; s[1] += rh[i] * rh[i]; s[2] += g1[i] * g1[i];
        s[3] += xv[i] * xv[i]; s[4] += hv[i] * hv[i];
    }
    breduce<5>(s, sdata);
    const float swh = s[0], srh = s[1], sux = s[2], sx = s[3], sh = s[4];
    const float nW = fmaxf(sqrtf(swh), 1e-15f), nrh = fmaxf(sqrtf(srh), 1e-15f);
    const float nU = fmaxf(sqrtf(sux), 1e-15f), nx = fmaxf(sqrtf(sx), 1e-15f);
    const float fW = (swh == 0.f) ? 0.f : tanhf(nW / nrh * artanh_c(nrh)) / nW;
    const float fU = (sux == 0.f) ? 0.f : tanhf(nU / nx * artanh_c(nx)) / nU;

    float Wv[4], Uv[4], pxy[1] = {0.f};
#pragma unroll
    for (int i = 0; i < 4; ++i) {
        Wv[i] = fW * g0[i]; Uv[i] = fU * g1[i];
        pxy[0] += Wv[i] * Uv[i];
    }
    breduce<1>(pxy, sdata);
    float x2 = fW * fW * swh, y2 = fU * fU * sux, xy = pxy[0];
    float den = fmaxf(1.f + 2.f * xy + x2 * y2, 1e-15f);
    float ca = (1.f + 2.f * xy + y2) / den, cb = (1.f - x2) / den;

    float t1[4], sb[3] = {0.f, 0.f, 0.f};
#pragma unroll
    for (int i = 0; i < 4; ++i) {
        t1[i] = ca * Wv[i] + cb * Uv[i];
        sb[0] += t1[i] * t1[i]; sb[1] += bv[i] * bv[i]; sb[2] += t1[i] * bv[i];
    }
    breduce<3>(sb, sdata);
    den = fmaxf(1.f + 2.f * sb[2] + sb[0] * sb[1], 1e-15f);
    ca = (1.f + 2.f * sb[2] + sb[1]) / den; cb = (1.f - sb[0]) / den;

    float ht[4], sd[2] = {0.f, 0.f};
#pragma unroll
    for (int i = 0; i < 4; ++i) {
        ht[i] = ca * t1[i] + cb * bv[i];
        sd[0] += ht[i] * ht[i]; sd[1] += hv[i] * ht[i];
    }
    breduce<2>(sd, sdata);
    // delta = mobius_add(-h, h_tilde)
    const float y2d = sd[0], xyd = -sd[1];
    float dend = fmaxf(1.f + 2.f * xyd + sh * y2d, 1e-15f);
    float cad = (1.f + 2.f * xyd + y2d) / dend, cbd = (1.f - sh) / dend;

    float dl[4], sm[2] = {0.f, 0.f};
#pragma unroll
    for (int i = 0; i < 4; ++i) {
        dl[i] = cad * (-hv[i]) + cbd * ht[i];
        float m = zv[i] * dl[i];
        sm[0] += m * m; sm[1] += dl[i] * dl[i];
    }
    breduce<2>(sm, sdata);
    const float nm = fmaxf(sqrtf(sm[0]), 1e-15f), nd = fmaxf(sqrtf(sm[1]), 1e-15f);
    const float fM = (sm[0] == 0.f) ? 0.f : tanhf(nm / nd * artanh_c(nd)) / nm;

    float pm[4], shp[1] = {0.f};
#pragma unroll
    for (int i = 0; i < 4; ++i) {
        pm[i] = fM * (zv[i] * dl[i]);
        shp[0] += hv[i] * pm[i];
    }
    breduce<1>(shp, sdata);
    // h_out = mobius_add(h, pm)
    const float y2o = fM * fM * sm[0], xyo = shp[0];
    float deno = fmaxf(1.f + 2.f * xyo + sh * y2o, 1e-15f);
    float cao = (1.f + 2.f * xyo + y2o) / deno, cbo = (1.f - sh) / deno;
#pragma unroll
    for (int i = 0; i < 4; ++i)
        out[base + t + i * 256] = cao * hv[i] + cbo * pm[i];
}

// ---------------------------------------------------------------------------
extern "C" void kernel_launch(void* const* d_in, const int* in_sizes, int n_in,
                              void* d_out, int out_size, void* d_ws, size_t ws_size,
                              hipStream_t stream) {
    const float* hyperx = (const float*)d_in[0];   // (16384, 1024)
    const float* hyperh = (const float*)d_in[1];   // (16384, 1024)
    const float* w_ih   = (const float*)d_in[2];   // (3072, 1024)
    const float* w_hh   = (const float*)d_in[3];   // (3072, 1024)
    const float* bias   = (const float*)d_in[4];   // (3, 1024): [r, h, z]
    float* out = (float*)d_out;

    const size_t H = HDIM, N = NROWS;
    const size_t NH = N * H;       // 16,777,216
    const size_t HH = H * H;       // 1,048,576
    char* ws = (char*)d_ws;
    // workspace layout (300 MB total, buffers reused across phases)
    unsigned short* XB   = (unsigned short*)(ws);                         // x bf16
    unsigned short* HB   = (unsigned short*)(ws + 2 * NH);                // h bf16
    unsigned short* WIHB = (unsigned short*)(ws + 4 * NH);                // w_ih bf16
    unsigned short* WHHB = (unsigned short*)(ws + 4 * NH + 6 * HH);       // w_hh bf16
    float*          G0   = (float*)(ws + 4 * NH + 12 * HH);               // gemm out
    float*          G1   = (float*)(ws + 8 * NH + 12 * HH);               // gemm out
    float*          ZT   = (float*)(ws + 12 * NH + 12 * HH);              // z_t fp32
    unsigned short* RHB  = (unsigned short*)(ws + 16 * NH + 12 * HH);     // rh bf16

    // 1) fp32 -> bf16 conversions
    cvt_bf16_kernel<<<2048, 256, 0, stream>>>(hyperx, XB, (int)NH);
    cvt_bf16_kernel<<<2048, 256, 0, stream>>>(hyperh, HB, (int)NH);
    cvt_bf16_kernel<<<512, 256, 0, stream>>>(w_ih, WIHB, (int)(3 * HH));
    cvt_bf16_kernel<<<512, 256, 0, stream>>>(w_hh, WHHB, (int)(3 * HH));

    dim3 gg(8, 128);  // 1024 cols / 128, 16384 rows / 128

    // 2) z gate: G0 = h @ W_hz^T (whh rows 2H..3H), G1 = x @ W_iz^T
    gemm_bf16_wmma<<<gg, 256, 0, stream>>>(HB, WHHB + 2 * HH, G0, (int)H, (int)H);
    gemm_bf16_wmma<<<gg, 256, 0, stream>>>(XB, WIHB + 2 * HH, G1, (int)H, (int)H);
    gate_kernel<<<NROWS, 256, 0, stream>>>(G0, G1, hyperh, hyperx,
                                           bias + 2 * H, ZT, RHB, 0);

    // 3) r gate: G0 = h @ W_hr^T (rows 0..H), G1 = x @ W_ir^T; fuses rh_t
    gemm_bf16_wmma<<<gg, 256, 0, stream>>>(HB, WHHB, G0, (int)H, (int)H);
    gemm_bf16_wmma<<<gg, 256, 0, stream>>>(XB, WIHB, G1, (int)H, (int)H);
    gate_kernel<<<NROWS, 256, 0, stream>>>(G0, G1, hyperh, hyperx,
                                           bias, ZT, RHB, 1);

    // 4) h_tilde GEMMs: G0 = rh @ W_hh^T (rows H..2H), G1 = x @ W_ih^T
    gemm_bf16_wmma<<<gg, 256, 0, stream>>>(RHB, WHHB + HH, G0, (int)H, (int)H);
    gemm_bf16_wmma<<<gg, 256, 0, stream>>>(XB, WIHB + HH, G1, (int)H, (int)H);

    // 5) final: h_tilde, delta_h, h_out
    final_kernel<<<NROWS, 256, 0, stream>>>(G0, G1, RHB, hyperx, hyperh,
                                            ZT, bias + H, out);
    (void)in_sizes; (void)n_in; (void)out_size; (void)ws_size;
}
